// DeformableTransformerEncoderLayer_63531156242626
// MI455X (gfx1250) — compile-verified, matrix operand order checked by reading, not run on hardware
//
#include <hip/hip_runtime.h>

// ---------------------------------------------------------------------------
// Types for CDNA5 WMMA
// ---------------------------------------------------------------------------
typedef __attribute__((ext_vector_type(16))) __bf16 v16bf;
typedef __attribute__((ext_vector_type(8)))  float  v8f;

__device__ __forceinline__ unsigned bf_bits(float f) {
  // round-to-nearest-even f32 -> bf16 bits (low 16)
  unsigned u = __builtin_bit_cast(unsigned, f);
  u += 0x7FFFu + ((u >> 16) & 1u);
  return u >> 16;
}
__device__ __forceinline__ unsigned pack_bf2(float lo, float hi) {
  return bf_bits(lo) | (bf_bits(hi) << 16);
}

struct Frag256 { uint4 lo; uint4 hi; };
__device__ __forceinline__ v16bf frag_pack(uint4 lo, uint4 hi) {
  Frag256 t{lo, hi};
  return __builtin_bit_cast(v16bf, t);
}

// ---------------------------------------------------------------------------
// GEMM: C[M,N] = (A (+A2)) @ Bw^T + bias   (Bw is torch [N,K] layout)
// bf16 WMMA 16x16x32, f32 accumulation. Block = 256 thr = 8 waves.
// Block tile 128(M) x 32(N), BK=32. Wave w handles rows [w*16, w*16+16),
// both 16-wide N tiles (2 accumulators, A-fragment reuse).
// LDS layout is fragment-native: As[row][k] (64B rows), Bs[n][k] (64B cols),
// so each WMMA operand is exactly two ds_load_b128.
// ---------------------------------------------------------------------------
#define BM 128
#define BN 32
#define BK 32

template <bool HAS_A2, bool RELU>
__global__ __launch_bounds__(256) void gemm_bf16_wmma(
    const float* __restrict__ A, const float* __restrict__ A2, int lda,
    const float* __restrict__ Bw, const float* __restrict__ bias,
    float* __restrict__ C, int ldc, int M, int N, int K)
{
  __shared__ unsigned As[BM * BK / 2];  // 2048 dwords = 8 KB, [row][kpair]
  __shared__ unsigned Bs[BN * BK / 2];  //  512 dwords = 2 KB, [n][kpair]

  const int tid  = threadIdx.x;
  const int lane = tid & 31;
  const int wave = tid >> 5;            // 8 m-tiles
  const int mblk = blockIdx.x * BM;
  const int nblk = blockIdx.y * BN;

  v8f acc0 = {0.f, 0.f, 0.f, 0.f, 0.f, 0.f, 0.f, 0.f};
  v8f acc1 = {0.f, 0.f, 0.f, 0.f, 0.f, 0.f, 0.f, 0.f};

  for (int k0 = 0; k0 < K; k0 += BK) {
    // ---- stage A tile (128 rows x 32 k), f32 -> packed bf16x2, branchless ----
#pragma unroll
    for (int e = 0; e < 8; ++e) {
      const int uidx = e * 256 + tid;
      const int r  = uidx >> 4;           // row in tile
      const int kp = (uidx & 15) * 2;     // k of pair
      const int grow = mblk + r;
      const int gr = grow < M ? grow : (M - 1);
      const float* pa = A + (size_t)gr * lda + k0 + kp;
      float v0 = pa[0], v1 = pa[1];
      if (HAS_A2) {
        const float* p2 = A2 + (size_t)gr * lda + k0 + kp;
        v0 += p2[0];
        v1 += p2[1];
      }
      if (grow >= M) { v0 = 0.f; v1 = 0.f; }
      As[uidx] = pack_bf2(v0, v1);
    }
    // ---- stage B tile (32 n x 32 k), k-contiguous per n ----
#pragma unroll
    for (int e = 0; e < 2; ++e) {
      const int uidx = e * 256 + tid;
      const int n  = uidx >> 4;
      const int kp = (uidx & 15) * 2;
      const float* pb = Bw + (size_t)(nblk + n) * K + k0 + kp;
      Bs[uidx] = pack_bf2(pb[0], pb[1]);
    }
    // prefetch next A panel (gfx1250 global_prefetch path)
    if (k0 + BK < K) {
      const int prow = mblk + tid / 2;
      if (prow < M)
        __builtin_prefetch(A + (size_t)prow * lda + k0 + BK + (tid & 1) * 16, 0, 0);
    }
    __syncthreads();

    // ---- fragments: two contiguous 16B runs per lane (ISA 16-bit layout) ----
    // lanes 0-15: kb=0  -> ks {0..7, 16..23};  lanes 16-31: kb=8 -> {8..15, 24..31}
    const int kbu = (lane < 16) ? 0 : 4;        // dword offset of k-run
    const int row = wave * 16 + (lane & 15);
    const uint4 alo = *(const uint4*)&As[row * 16 + kbu];
    const uint4 ahi = *(const uint4*)&As[row * 16 + 8 + kbu];
    const v16bf afrag = frag_pack(alo, ahi);

    const int col = lane & 15;
    const uint4 b0lo = *(const uint4*)&Bs[col * 16 + kbu];
    const uint4 b0hi = *(const uint4*)&Bs[col * 16 + 8 + kbu];
    const uint4 b1lo = *(const uint4*)&Bs[(16 + col) * 16 + kbu];
    const uint4 b1hi = *(const uint4*)&Bs[(16 + col) * 16 + 8 + kbu];

    acc0 = __builtin_amdgcn_wmma_f32_16x16x32_bf16(
        false, afrag, false, frag_pack(b0lo, b0hi), (short)0, acc0, false, false);
    acc1 = __builtin_amdgcn_wmma_f32_16x16x32_bf16(
        false, afrag, false, frag_pack(b1lo, b1hi), (short)0, acc1, false, false);
    __syncthreads();
  }

  // ---- epilogue: C/D layout: VGPR r -> (M = r or r+8, N = lane&15) ----
  const int nb0   = nblk + (lane & 15);
  const int mrow0 = mblk + wave * 16 + ((lane & 16) ? 8 : 0);
  const float bv0 = bias[nb0];
  const float bv1 = bias[nb0 + 16];
#pragma unroll
  for (int r = 0; r < 8; ++r) {
    const int rowg = mrow0 + r;
    if (rowg < M) {
      float v0 = acc0[r] + bv0;
      float v1 = acc1[r] + bv1;
      if (RELU) {
        v0 = v0 > 0.f ? v0 : 0.f;
        v1 = v1 > 0.f ? v1 : 0.f;
      }
      C[(size_t)rowg * ldc + nb0]      = v0;
      C[(size_t)rowg * ldc + nb0 + 16] = v1;
    }
  }
}

// ---------------------------------------------------------------------------
// Softmax over 16 elements per (query, head), in place
// ---------------------------------------------------------------------------
__global__ __launch_bounds__(256) void softmax16_kernel(float* __restrict__ a, int n)
{
  int i = blockIdx.x * blockDim.x + threadIdx.x;
  if (i >= n) return;
  float* p = a + (size_t)i * 16;
  float v[16], m = -3.4e38f;
#pragma unroll
  for (int j = 0; j < 16; ++j) { v[j] = p[j]; m = fmaxf(m, v[j]); }
  float s = 0.f;
#pragma unroll
  for (int j = 0; j < 16; ++j) { v[j] = __expf(v[j] - m); s += v[j]; }
  float inv = 1.f / s;
#pragma unroll
  for (int j = 0; j < 16; ++j) p[j] = v[j] * inv;
}

// ---------------------------------------------------------------------------
// MSDA bilinear gather. One block (384 threads) per query; tx = h*48 + c.
// ---------------------------------------------------------------------------
__global__ __launch_bounds__(384) void msda_kernel(
    const float* __restrict__ value,  // [Lq,384]  (h*48+c)
    const float* __restrict__ offs,   // [Lq,256]  (h,l,p,xy)
    const float* __restrict__ attn,   // [Lq,128]  (h,l*4+p)
    const float* __restrict__ refp,   // [Lq,4,2]
    float* __restrict__ out, int Lq)
{
  const int q = blockIdx.x;
  const int tx = threadIdx.x;
  const int h = tx / 48, c = tx % 48;
  const int lvlH[4] = {100, 50, 25, 13};
  const int lvlW[4] = {134, 67, 34, 17};
  const int lvlS[4] = {0, 13400, 16750, 17600};

  const float* offq = offs + (size_t)q * 256 + h * 32;
  const float* attq = attn + (size_t)q * 128 + h * 16;
  const float* refq = refp + (size_t)q * 8;

  float sum = 0.f;
#pragma unroll
  for (int l = 0; l < 4; ++l) {
    const int Hl = lvlH[l], Wl = lvlW[l], st = lvlS[l];
    const float rx = refq[l * 2 + 0], ry = refq[l * 2 + 1];
#pragma unroll
    for (int p = 0; p < 4; ++p) {
      const float ox = offq[l * 8 + p * 2 + 0];
      const float oy = offq[l * 8 + p * 2 + 1];
      const float x = (rx + ox / (float)Wl) * (float)Wl - 0.5f;
      const float y = (ry + oy / (float)Hl) * (float)Hl - 0.5f;
      const float xf = floorf(x), yf = floorf(y);
      const float dx = x - xf, dy = y - yf;
      const int x0 = (int)xf, y0 = (int)yf;
      float acc = 0.f;
#pragma unroll
      for (int oyi = 0; oyi < 2; ++oyi) {
#pragma unroll
        for (int oxi = 0; oxi < 2; ++oxi) {
          const int xi = x0 + oxi, yi = y0 + oyi;
          if (xi >= 0 && xi < Wl && yi >= 0 && yi < Hl) {
            const float w = (oxi ? dx : 1.f - dx) * (oyi ? dy : 1.f - dy);
            acc += w * value[(size_t)(st + yi * Wl + xi) * 384 + h * 48 + c];
          }
        }
      }
      sum += attq[l * 4 + p] * acc;
    }
  }
  out[(size_t)q * 384 + tx] = sum;
}

// ---------------------------------------------------------------------------
// LayerNorm(a_row + b_row) over `width` (128 or 256); one wave32 per row.
// ---------------------------------------------------------------------------
__global__ __launch_bounds__(256) void ln_residual_kernel(
    const float* __restrict__ a, int lda,
    const float* __restrict__ b, int ldb,
    const float* __restrict__ g, const float* __restrict__ be,
    float* __restrict__ out, int ldo,
    int rows, int width)
{
  const int wave = threadIdx.x >> 5, lane = threadIdx.x & 31;
  const int row = blockIdx.x * 8 + wave;
  if (row >= rows) return;
  const float* pa = a + (size_t)row * lda;
  const float* pb = b + (size_t)row * ldb;
  const int per = width >> 5;  // 4 or 8
  float vals[8];
  float s = 0.f;
  for (int j = 0; j < per; ++j) {
    int cidx = lane + j * 32;
    float v = pa[cidx] + pb[cidx];
    vals[j] = v;
    s += v;
  }
  for (int m = 16; m > 0; m >>= 1) s += __shfl_xor(s, m, 32);
  const float mean = s / (float)width;
  float vs = 0.f;
  for (int j = 0; j < per; ++j) { float d = vals[j] - mean; vs += d * d; }
  for (int m = 16; m > 0; m >>= 1) vs += __shfl_xor(vs, m, 32);
  const float inv = rsqrtf(vs / (float)width + 1e-5f);
  float* po = out + (size_t)row * ldo;
  for (int j = 0; j < per; ++j) {
    int cidx = lane + j * 32;
    po[cidx] = (vals[j] - mean) * inv * g[cidx] + be[cidx];
  }
}

// ---------------------------------------------------------------------------
// Host-side orchestration
// ---------------------------------------------------------------------------
extern "C" void kernel_launch(void* const* d_in, const int* in_sizes, int n_in,
                              void* d_out, int out_size, void* d_ws, size_t ws_size,
                              hipStream_t stream) {
  (void)in_sizes; (void)n_in; (void)out_size; (void)ws_size;
  const float* src    = (const float*)d_in[0];
  const float* pos    = (const float*)d_in[3];
  const float* refpts = (const float*)d_in[4];
  const float* W_off  = (const float*)d_in[8];
  const float* b_off  = (const float*)d_in[9];
  const float* W_attn = (const float*)d_in[10];
  const float* b_attn = (const float*)d_in[11];
  const float* W_val  = (const float*)d_in[12];
  const float* b_val  = (const float*)d_in[13];
  const float* W_out  = (const float*)d_in[14];
  const float* b_out  = (const float*)d_in[15];
  const float* g1xy   = (const float*)d_in[16];
  const float* b1xy   = (const float*)d_in[17];
  const float* g1zd   = (const float*)d_in[18];
  const float* b1zd   = (const float*)d_in[19];
  const float* fxy_w1 = (const float*)d_in[20];
  const float* fxy_b1 = (const float*)d_in[21];
  const float* fxy_w2 = (const float*)d_in[22];
  const float* fxy_b2 = (const float*)d_in[23];
  const float* fxy_g  = (const float*)d_in[24];
  const float* fxy_b  = (const float*)d_in[25];
  const float* fzd_w1 = (const float*)d_in[26];
  const float* fzd_b1 = (const float*)d_in[27];
  const float* fzd_w2 = (const float*)d_in[28];
  const float* fzd_b2 = (const float*)d_in[29];
  const float* fzd_g  = (const float*)d_in[30];
  const float* fzd_b  = (const float*)d_in[31];

  const int Lq = 17821;
  const size_t SZ_D  = (size_t)Lq * 384 * 4;
  const size_t SZ_XY = (size_t)Lq * 256 * 4;
  const size_t SZ_ZD = (size_t)Lq * 128 * 4;
  char* ws = (char*)d_ws;
  float* value = (float*)(ws);                             // R0, later x
  float* offb  = (float*)(ws + SZ_D);                      // R1, later hxy
  float* attnb = (float*)(ws + SZ_D + SZ_XY);              // R2
  float* msda  = (float*)(ws + SZ_D + SZ_XY + SZ_ZD);      // R3, later t2xy
  float* src2  = (float*)(ws + 2 * SZ_D + SZ_XY + SZ_ZD);  // R4, later hzd/t2zd
  float* x     = value;
  float* hxy   = offb;
  float* t2xy  = msda;
  float* hzd   = src2;
  float* t2zd  = (float*)((char*)src2 + SZ_ZD);
  float* out   = (float*)d_out;

  auto ggrid = [](int M, int N) { return dim3((unsigned)((M + BM - 1) / BM), (unsigned)(N / BN)); };
  const dim3 blk(256);

  // 1-3: projections (value from src; off/attn from q = src + pos)
  gemm_bf16_wmma<false, false><<<ggrid(Lq, 384), blk, 0, stream>>>(src, nullptr, 384, W_val,  b_val,  value, 384, Lq, 384, 384);
  gemm_bf16_wmma<true,  false><<<ggrid(Lq, 256), blk, 0, stream>>>(src, pos,     384, W_off,  b_off,  offb,  256, Lq, 256, 384);
  gemm_bf16_wmma<true,  false><<<ggrid(Lq, 128), blk, 0, stream>>>(src, pos,     384, W_attn, b_attn, attnb, 128, Lq, 128, 384);

  // 4: softmax over 16 per (q,h)
  softmax16_kernel<<<(Lq * 8 + 255) / 256, blk, 0, stream>>>(attnb, Lq * 8);

  // 5: deformable gather
  msda_kernel<<<Lq, 384, 0, stream>>>(value, offb, attnb, refpts, msda, Lq);

  // 6: output projection
  gemm_bf16_wmma<false, false><<<ggrid(Lq, 384), blk, 0, stream>>>(msda, nullptr, 384, W_out, b_out, src2, 384, Lq, 384, 384);

  // 7-8: LN1 split (residual with src)
  ln_residual_kernel<<<(Lq + 7) / 8, blk, 0, stream>>>(src,       384, src2,       384, g1xy, b1xy, x,       384, Lq, 256);
  ln_residual_kernel<<<(Lq + 7) / 8, blk, 0, stream>>>(src + 256, 384, src2 + 256, 384, g1zd, b1zd, x + 256, 384, Lq, 128);

  // 9-11: FFN xy (relu fused into first GEMM), final LN -> out[:, :256]
  gemm_bf16_wmma<false, true ><<<ggrid(Lq, 256), blk, 0, stream>>>(x,   nullptr, 384, fxy_w1, fxy_b1, hxy,  256, Lq, 256, 256);
  gemm_bf16_wmma<false, false><<<ggrid(Lq, 256), blk, 0, stream>>>(hxy, nullptr, 256, fxy_w2, fxy_b2, t2xy, 256, Lq, 256, 256);
  ln_residual_kernel<<<(Lq + 7) / 8, blk, 0, stream>>>(x, 384, t2xy, 256, fxy_g, fxy_b, out, 384, Lq, 256);

  // 12-14: FFN zd, final LN -> out[:, 256:]
  gemm_bf16_wmma<false, true ><<<ggrid(Lq, 128), blk, 0, stream>>>(x + 256, nullptr, 384, fzd_w1, fzd_b1, hzd,  128, Lq, 128, 128);
  gemm_bf16_wmma<false, false><<<ggrid(Lq, 128), blk, 0, stream>>>(hzd,     nullptr, 128, fzd_w2, fzd_b2, t2zd, 128, Lq, 128, 128);
  ln_residual_kernel<<<(Lq + 7) / 8, blk, 0, stream>>>(x + 256, 384, t2zd, 128, fzd_g, fzd_b, out + 256, 384, Lq, 128);
}
